// MultiheadAttention_64673617543186
// MI455X (gfx1250) — matrix operand order repaired and checked by reading, not compiled, not run on hardware
//
#include <hip/hip_runtime.h>
#include <cstdint>

typedef __bf16 bf16_t;
typedef __attribute__((ext_vector_type(16))) __bf16 v16bf;
typedef __attribute__((ext_vector_type(8)))  float  v8f;

struct BF16Frag { uint4 lo, hi; };

#define LOG2E 1.44269504088896340736f

constexpr int kB = 2, kS = 2048, kE = 1024, kH = 16, kD = 64;
constexpr int kBH = kB * kH;                       // 32
constexpr size_t kQKVsz = (size_t)kBH * kS * kD;   // elems per q/k/v buffer

__device__ __forceinline__ v8f wmma_bf16(v16bf a, v16bf b, v8f c) {
  return __builtin_amdgcn_wmma_f32_16x16x32_bf16(false, a, false, b, (short)0, c,
                                                 false, false);
}

// CDNA5 async global->LDS copy, 16B/lane; ISA adds INST_OFFSET to BOTH the
// LDS address and the global address, so one base covers a whole row.
#define ASYNC_B128(gptr, ldsaddr, OFF)                                        \
  asm volatile("global_load_async_to_lds_b128 %0, %1, off offset:" #OFF       \
               :: "v"(ldsaddr), "v"(gptr) : "memory")

__device__ __forceinline__ void s_wait_async() {
  asm volatile("s_wait_asynccnt 0x0" ::: "memory");
}
__device__ __forceinline__ uint32_t lds_addr(const void* p) {
  return (uint32_t)(uintptr_t)p;
}

// A fragment (16 rows x 32 K), source row-major: row = M, contiguous K.
__device__ __forceinline__ v16bf load_A_frag(const bf16_t* base, int rowStride,
                                             int lane) {
  int row = lane & 15;
  int k0  = (lane >> 4) * 16;
  const bf16_t* p = base + row * rowStride + k0;
  BF16Frag f;
  f.lo = *(const uint4*)(p);
  f.hi = *(const uint4*)(p + 8);
  return __builtin_bit_cast(v16bf, f);
}

// B fragment (32 K x 16 N), source row-major with row = N, contiguous K.
__device__ __forceinline__ v16bf load_B_frag(const bf16_t* base, int rowStride,
                                             int lane) {
  int n  = lane & 15;
  int k0 = (lane >> 4) * 16;
  const bf16_t* p = base + n * rowStride + k0;
  BF16Frag f;
  f.lo = *(const uint4*)(p);
  f.hi = *(const uint4*)(p + 8);
  return __builtin_bit_cast(v16bf, f);
}

__global__ void cvt_f32_bf16(const float* __restrict__ in,
                             bf16_t* __restrict__ out, int n) {
  int i = (blockIdx.x * blockDim.x + threadIdx.x) * 4;
  if (i >= n) return;
  float4 f = *(const float4*)(in + i);
  union { bf16_t e[4]; uint2 u; } t;
  t.e[0] = (bf16_t)f.x; t.e[1] = (bf16_t)f.y;
  t.e[2] = (bf16_t)f.z; t.e[3] = (bf16_t)f.w;
  *(uint2*)(out + i) = t.u;
}

// C = A[MxK] * W[NxK]^T + bias.  Block 128x256, BK=64, double-buffered async
// LDS staging, 8 waves each computing a 64x64 tile (32 WMMA / stage / wave).
// MODE 0: qkv epilogue (bias, q-scale, head scatter, bf16). MODE 1: fp32 out.
template <int MODE>
__global__ __launch_bounds__(256) void gemm_bf16_kernel(
    const bf16_t* __restrict__ A, const bf16_t* __restrict__ W,
    const float* __restrict__ bias, void* __restrict__ Out,
    int M, int N, int K) {
  constexpr int BM = 128, BN = 256, BK = 64;
  __shared__ bf16_t As[2 * BM * BK];   // 32 KB
  __shared__ bf16_t Bs[2 * BN * BK];   // 64 KB
  const int tid = threadIdx.x, lane = tid & 31, wave = tid >> 5;
  const int wm = wave >> 2;   // 0..1 : 64-row band
  const int wn = wave & 3;    // 0..3 : 64-col band
  const int bx = blockIdx.x, by = blockIdx.y;
  v8f acc[4][4] = {};

  const int ra = tid >> 1, ca = (tid & 1) * 32;   // A staging: 32 elems/thread
  auto prefetch = [&](int kk, int pb) {
    const bf16_t* ga = A + (size_t)(by * BM + ra) * K + kk + ca;
    uint32_t la = lds_addr(&As[pb * BM * BK + ra * BK + ca]);
    ASYNC_B128(ga, la, 0);  ASYNC_B128(ga, la, 16);
    ASYNC_B128(ga, la, 32); ASYNC_B128(ga, la, 48);
    const bf16_t* gb = W + (size_t)(bx * BN + tid) * K + kk;  // 64 elems/thread
    uint32_t lb = lds_addr(&Bs[pb * BN * BK + tid * BK]);
    ASYNC_B128(gb, lb, 0);   ASYNC_B128(gb, lb, 16);
    ASYNC_B128(gb, lb, 32);  ASYNC_B128(gb, lb, 48);
    ASYNC_B128(gb, lb, 64);  ASYNC_B128(gb, lb, 80);
    ASYNC_B128(gb, lb, 96);  ASYNC_B128(gb, lb, 112);
  };

  int buf = 0;
  prefetch(0, 0);
  for (int k0 = 0; k0 < K; k0 += BK) {
    s_wait_async();
    __syncthreads();
    if (k0 + BK < K) prefetch(k0 + BK, buf ^ 1);
    const bf16_t* as = As + buf * BM * BK;
    const bf16_t* bs = Bs + buf * BN * BK;
#pragma unroll
    for (int ks = 0; ks < 2; ++ks) {
      v16bf bfr[4];
#pragma unroll
      for (int j = 0; j < 4; ++j)
        bfr[j] = load_B_frag(bs + (wn * 64 + j * 16) * BK + ks * 32, BK, lane);
#pragma unroll
      for (int i = 0; i < 4; ++i) {
        v16bf a = load_A_frag(as + (wm * 64 + i * 16) * BK + ks * 32, BK, lane);
#pragma unroll
        for (int j = 0; j < 4; ++j)
          acc[i][j] = wmma_bf16(a, bfr[j], acc[i][j]);
      }
    }
    buf ^= 1;
  }

  const int half = lane >> 4, colL = lane & 15;
#pragma unroll
  for (int i = 0; i < 4; ++i) {
#pragma unroll
    for (int j = 0; j < 4; ++j) {
      int col = bx * BN + wn * 64 + j * 16 + colL;
      float bv = bias[col];
#pragma unroll
      for (int rr = 0; rr < 8; ++rr) {
        int m = by * BM + wm * 64 + i * 16 + half * 8 + rr;
        float v = acc[i][j][rr] + bv;
        if constexpr (MODE == 0) {
          int which = col >> 10, e = col & 1023;
          int h = e >> 6, d = e & 63;
          if (which == 0) v *= 0.125f;  // 1/sqrt(D)
          int b = m >> 11, s = m & 2047;
          ((bf16_t*)Out)[(size_t)which * kQKVsz +
                         (((size_t)(b * kH + h)) * kS + s) * kD + d] = (bf16_t)v;
        } else {
          ((float*)Out)[(size_t)m * N + col] = v;
        }
      }
    }
  }
}

// Flash attention: one WG (8 waves, 256 thr) = 128 query rows of one (b,h).
__global__ __launch_bounds__(256) void attn_kernel(
    const bf16_t* __restrict__ Q, const bf16_t* __restrict__ Kb,
    const bf16_t* __restrict__ Vb, const float* __restrict__ mask,
    bf16_t* __restrict__ Y) {
  __shared__ bf16_t Qs[128 * 64];   // 16 KB
  __shared__ bf16_t Ks[64 * 64];    //  8 KB
  __shared__ bf16_t Vs[64 * 64];    //  8 KB, transposed: Vs[d*64 + kv]
  __shared__ float  Sc[128 * 64];   // 32 KB
  __shared__ bf16_t Ps[128 * 64];   // 16 KB
  __shared__ float  m_s[128], l_s[128], f_s[128], red[256];

  const int tid = threadIdx.x, lane = tid & 31, wave = tid >> 5;
  const int qc = blockIdx.x & 15;
  const int bh = blockIdx.x >> 4;
  const int b = bh >> 4, h = bh & 15;
  const int q0 = qc * 128;
  const int r = tid >> 1, c32 = (tid & 1) * 32;    // 128 rows x 2 halves
  const int rk = tid >> 2, ck = (tid & 3) * 16;    // 64 rows x 4 quarters

  {  // Q tile (128x64), async
    const bf16_t* g = Q + ((size_t)bh * kS + q0 + r) * kD + c32;
    uint32_t l = lds_addr(&Qs[r * 64 + c32]);
    ASYNC_B128(g, l, 0);  ASYNC_B128(g, l, 16);
    ASYNC_B128(g, l, 32); ASYNC_B128(g, l, 48);
  }
  if (tid < 128) { m_s[tid] = -3.0e38f; l_s[tid] = 0.0f; }
  s_wait_async();
  __syncthreads();
  const v16bf aq0 = load_A_frag(&Qs[(wave * 16) * 64 + 0],  64, lane);
  const v16bf aq1 = load_A_frag(&Qs[(wave * 16) * 64 + 32], 64, lane);

  v8f o[4] = {};

  for (int kc = 0; kc < kS; kc += 64) {
    __syncthreads();  // all reads of Ks/Vs/Ps from previous iter are done
    {  // K tile (64x64), async
      const bf16_t* g = Kb + ((size_t)bh * kS + kc + rk) * kD + ck;
      uint32_t l = lds_addr(&Ks[rk * 64 + ck]);
      ASYNC_B128(g, l, 0); ASYNC_B128(g, l, 16);
    }
    {  // V tile transposed into Vs[d][kv] (VGPR-staged)
      int kv = rk, d0 = ck;
      union { uint4 u[2]; bf16_t e[16]; } bufv;
      const uint4* g = (const uint4*)(Vb + ((size_t)bh * kS + kc + kv) * kD + d0);
      bufv.u[0] = g[0]; bufv.u[1] = g[1];
#pragma unroll
      for (int i = 0; i < 16; ++i) Vs[(d0 + i) * 64 + kv] = bufv.e[i];
    }
    s_wait_async();
    __syncthreads();

    // S = Q*K^T : 16 rows per wave x 64 kv
#pragma unroll
    for (int j = 0; j < 4; ++j) {
      v8f sc = {};
      v16bf bk0 = load_B_frag(&Ks[(j * 16) * 64 + 0],  64, lane);
      v16bf bk1 = load_B_frag(&Ks[(j * 16) * 64 + 32], 64, lane);
      sc = wmma_bf16(aq0, bk0, sc);
      sc = wmma_bf16(aq1, bk1, sc);
      int col = j * 16 + (lane & 15);
      int r0  = wave * 16 + (lane >> 4) * 8;
#pragma unroll
      for (int rr = 0; rr < 8; ++rr) Sc[(r0 + rr) * 64 + col] = sc[rr];
    }
    __syncthreads();

    // Online softmax: 2 threads per row, 32 columns each.
    {
      const float* mrow = mask + ((size_t)b * kS + (q0 + r)) * kS + kc + c32;
      float* srow = &Sc[r * 64 + c32];
      float mx = -3.0e38f;
      for (int cc = 0; cc < 32; ++cc) {
        float sv = srow[cc] - 10000.0f * mrow[cc];
        srow[cc] = sv;
        mx = fmaxf(mx, sv);
      }
      red[tid] = mx;
    }
    __syncthreads();
    if (tid < 128) {
      float mx = fmaxf(m_s[tid], fmaxf(red[2 * tid], red[2 * tid + 1]));
      f_s[tid] = exp2f((m_s[tid] - mx) * LOG2E);
      m_s[tid] = mx;
    }
    __syncthreads();
    {
      float newm = m_s[r];
      float* srow = &Sc[r * 64 + c32];
      bf16_t* prow = &Ps[r * 64 + c32];
      float sum = 0.0f;
      for (int cc = 0; cc < 32; ++cc) {
        float p = exp2f((srow[cc] - newm) * LOG2E);
        prow[cc] = (bf16_t)p;
        sum += p;
      }
      red[tid] = sum;
    }
    __syncthreads();
    if (tid < 128)
      l_s[tid] = l_s[tid] * f_s[tid] + red[2 * tid] + red[2 * tid + 1];
    __syncthreads();

    {  // rescale accumulators
      int r0 = wave * 16 + (lane >> 4) * 8;
      float fr[8];
#pragma unroll
      for (int rr = 0; rr < 8; ++rr) fr[rr] = f_s[r0 + rr];
#pragma unroll
      for (int j = 0; j < 4; ++j)
#pragma unroll
        for (int rr = 0; rr < 8; ++rr) o[j][rr] *= fr[rr];
    }
    {  // O += P*V
      v16bf ap0 = load_A_frag(&Ps[(wave * 16) * 64 + 0],  64, lane);
      v16bf ap1 = load_A_frag(&Ps[(wave * 16) * 64 + 32], 64, lane);
#pragma unroll
      for (int j = 0; j < 4; ++j) {
        v16bf bv0 = load_B_frag(&Vs[(j * 16) * 64 + 0],  64, lane);
        v16bf bv1 = load_B_frag(&Vs[(j * 16) * 64 + 32], 64, lane);
        o[j] = wmma_bf16(ap0, bv0, o[j]);
        o[j] = wmma_bf16(ap1, bv1, o[j]);
      }
    }
  }

  {  // normalize and write Y[b*S+s][h*64+d] as bf16
    int colL = lane & 15, half = lane >> 4;
#pragma unroll
    for (int rr = 0; rr < 8; ++rr) {
      int rl = wave * 16 + half * 8 + rr;
      float inv = 1.0f / l_s[rl];
      size_t row = (size_t)b * kS + q0 + rl;
#pragma unroll
      for (int j = 0; j < 4; ++j)
        Y[row * kE + h * kD + j * 16 + colL] = (bf16_t)(o[j][rr] * inv);
    }
  }
}

extern "C" void kernel_launch(void* const* d_in, const int* in_sizes, int n_in,
                              void* d_out, int out_size, void* d_ws,
                              size_t ws_size, hipStream_t stream) {
  const float* x     = (const float*)d_in[0];
  const float* mask  = (const float*)d_in[1];
  const float* w_in  = (const float*)d_in[2];
  const float* b_in  = (const float*)d_in[3];
  const float* w_out = (const float*)d_in[4];
  const float* b_out = (const float*)d_in[5];

  char* ws = (char*)d_ws;
  bf16_t* xb  = (bf16_t*)(ws);                       // 8 MB  [4096 x 1024]
  bf16_t* wib = (bf16_t*)(ws + ((size_t)8  << 20));  // 6 MB  [3072 x 1024]
  bf16_t* wob = (bf16_t*)(ws + ((size_t)14 << 20));  // 2 MB  [1024 x 1024]
  bf16_t* qkv = (bf16_t*)(ws + ((size_t)16 << 20));  // 24 MB [3][32][2048][64]
  bf16_t* yb  = (bf16_t*)(ws + ((size_t)40 << 20));  // 8 MB  [4096 x 1024]

  const int nX  = kB * kS * kE;   // 4194304
  const int nWi = 3 * kE * kE;    // 3145728
  const int nWo = kE * kE;        // 1048576
  cvt_f32_bf16<<<nX  / 1024, 256, 0, stream>>>(x,     xb,  nX);
  cvt_f32_bf16<<<nWi / 1024, 256, 0, stream>>>(w_in,  wib, nWi);
  cvt_f32_bf16<<<nWo / 1024, 256, 0, stream>>>(w_out, wob, nWo);

  // QKV projection: M=4096, N=3072, K=1024
  gemm_bf16_kernel<0><<<dim3(3072 / 256, 4096 / 128), 256, 0, stream>>>(
      xb, wib, b_in, qkv, kB * kS, 3 * kE, kE);

  // Attention: 32 (b,h) x 16 query chunks of 128 rows
  attn_kernel<<<kBH * (kS / 128), 256, 0, stream>>>(
      qkv, qkv + kQKVsz, qkv + 2 * kQKVsz, mask, yb);

  // Output projection: M=4096, N=1024, K=1024 -> fp32 d_out
  gemm_bf16_kernel<1><<<dim3(1024 / 256, 4096 / 128), 256, 0, stream>>>(
      yb, wob, b_out, d_out, kB * kS, kE, kE);
}
